// DotProductAttention_5145370821132
// MI455X (gfx1250) — matrix operand order repaired and checked by reading, not compiled, not run on hardware
//
#include <hip/hip_runtime.h>
#include <hip/hip_bf16.h>

typedef __attribute__((ext_vector_type(16))) _Float16 v16h;
typedef __attribute__((ext_vector_type(8)))  _Float16 v8h;
typedef __attribute__((ext_vector_type(8)))  float    v8f;
typedef __attribute__((ext_vector_type(4)))  float    v4f;

#define B_  16
#define SQ_ 2048
#define SK_ 2048
#define D_  64

#define KSTR 72   // halfs per K row in LDS (144B: 16B-aligned, bank-spread)
#define VSTR 72   // halfs per transposed-V row (64 keys + 8 pad)
#define PSTR 72   // halfs per P row (64 keys + 8 pad)

#if __has_builtin(__builtin_amdgcn_exp2f)
#define EXP2(x) __builtin_amdgcn_exp2f(x)
#else
#define EXP2(x) exp2f(x)
#endif

static __device__ __forceinline__ v8f wmma16x16x32(v16h a, v16h b, v8f c) {
  return __builtin_amdgcn_wmma_f32_16x16x32_f16(false, a, false, b, (short)0, c,
                                                false, false);
}

// 16 contiguous halfs from LDS (two ds_load_b128) -> one f16 operand
static __device__ __forceinline__ v16h lds_read16(const _Float16* p) {
  v8h lo = *(const v8h*)p;
  v8h hi = *(const v8h*)(p + 8);
  v16h r;
#pragma unroll
  for (int j = 0; j < 8; ++j) { r[j] = lo[j]; r[8 + j] = hi[j]; }
  return r;
}

// max-reduce across each 16-lane half-wave, broadcast result (VALU-only DPP16)
static __device__ __forceinline__ float maxbcast16(float v) {
#if __has_builtin(__builtin_amdgcn_update_dpp)
  int x, y;
  x = __float_as_int(v);
  y = __builtin_amdgcn_update_dpp(x, x, 0xB1, 0xF, 0xF, false);  // quad_perm [1,0,3,2]
  v = fmaxf(v, __int_as_float(y));
  x = __float_as_int(v);
  y = __builtin_amdgcn_update_dpp(x, x, 0x4E, 0xF, 0xF, false);  // quad_perm [2,3,0,1]
  v = fmaxf(v, __int_as_float(y));
  x = __float_as_int(v);
  y = __builtin_amdgcn_update_dpp(x, x, 0x141, 0xF, 0xF, false); // row_half_mirror
  v = fmaxf(v, __int_as_float(y));
  x = __float_as_int(v);
  y = __builtin_amdgcn_update_dpp(x, x, 0x140, 0xF, 0xF, false); // row_mirror
  v = fmaxf(v, __int_as_float(y));
#else
#pragma unroll
  for (int m = 1; m < 16; m <<= 1) v = fmaxf(v, __shfl_xor(v, m, 32));
#endif
  return v;
}

__global__ __launch_bounds__(256)
void fa_wmma_kernel(const float* __restrict__ Qg, const float* __restrict__ Kg,
                    const float* __restrict__ Vg, const int* __restrict__ VLg,
                    float* __restrict__ Og) {
  __shared__ __align__(16) _Float16 sK [2 * 64 * KSTR];  // K chunk, f16 row-major, x2
  __shared__ __align__(16) _Float16 sVt[2 * 64 * VSTR];  // V chunk, f16 [d][k], x2
  __shared__ __align__(16) _Float16 sP [8 * 16 * PSTR];  // per-wave P bounce

  const int b    = blockIdx.x >> 4;    // 16 query-blocks per batch
  const int qblk = blockIdx.x & 15;
  const int wave = threadIdx.x >> 5;
  const int lane = threadIdx.x & 31;
  const int h    = lane >> 4;
  const int lm   = lane & 15;
  const int q0   = qblk * 128 + wave * 16;
  const int vlen = VLg[b];
  const int nch  = (vlen + 63) >> 6;   // 64-key chunks; rest contribute exactly 0

  // ---- Q tile -> f16 A-operands, pre-scaled by log2(e)/sqrt(D) ----
  // A operand k order chosen here (B matches): lane(row=lm, h): element e <-> K = 16h + e
  const float QSCALE = 0.125f * 1.44269504088896340736f;
  v16h Qa[2];
  {
    const float* qrow = Qg + ((size_t)b * SQ_ + q0 + lm) * D_;
#pragma unroll
    for (int c = 0; c < 2; ++c) {
      const int d0 = c * 32 + 16 * h;
      const int d1 = c * 32 + 16 * h + 8;
      v4f a0 = *(const v4f*)(qrow + d0);
      v4f a1 = *(const v4f*)(qrow + d0 + 4);
      v4f b0 = *(const v4f*)(qrow + d1);
      v4f b1 = *(const v4f*)(qrow + d1 + 4);
#pragma unroll
      for (int j = 0; j < 4; ++j) {
        Qa[c][j]      = (_Float16)(a0[j] * QSCALE);
        Qa[c][4 + j]  = (_Float16)(a1[j] * QSCALE);
        Qa[c][8 + j]  = (_Float16)(b0[j] * QSCALE);
        Qa[c][12 + j] = (_Float16)(b1[j] * QSCALE);
      }
    }
  }

  // all-ones B operand: P * ones -> row sums broadcast in C-layout
  v16h onesB;
#pragma unroll
  for (int j = 0; j < 16; ++j) onesB[j] = (_Float16)1.0f;

  v8f Oacc[4];
#pragma unroll
  for (int t = 0; t < 4; ++t) Oacc[t] = {};
  float mrow[8], lrow[8], corr8[8];
#pragma unroll
  for (int i = 0; i < 8; ++i) { mrow[i] = -1e30f; lrow[i] = 0.f; }

  const int ck = threadIdx.x >> 2;        // key within 64-chunk (0..63)
  const int cd = (threadIdx.x & 3) * 16;  // d offset (0,16,32,48)
  const size_t kvbase = ((size_t)b * SK_ + ck) * D_ + cd;

  // ---- prefetch chunk 0 into registers ----
  v4f kr[4], vr[4];
  {
    const float* kp = Kg + kvbase;
    const float* vp = Vg + kvbase;
#pragma unroll
    for (int j = 0; j < 4; ++j) {
      kr[j] = *(const v4f*)(kp + 4 * j);
      vr[j] = *(const v4f*)(vp + 4 * j);
    }
  }

  for (int kc = 0; kc < nch; ++kc) {
    const int kb = kc * 64;
    _Float16* bK  = sK  + (kc & 1) * (64 * KSTR);
    _Float16* bVt = sVt + (kc & 1) * (64 * VSTR);

    // ---- convert prefetched registers -> LDS (f16) ----
    {
      v8h ha, hb;
#pragma unroll
      for (int j = 0; j < 4; ++j) {
        ha[j] = (_Float16)kr[0][j]; ha[4 + j] = (_Float16)kr[1][j];
        hb[j] = (_Float16)kr[2][j]; hb[4 + j] = (_Float16)kr[3][j];
      }
      *(v8h*)(bK + ck * KSTR + cd)     = ha;  // ds_store_b128
      *(v8h*)(bK + ck * KSTR + cd + 8) = hb;
#pragma unroll
      for (int g = 0; g < 4; ++g)
#pragma unroll
        for (int j = 0; j < 4; ++j)
          bVt[(cd + g * 4 + j) * VSTR + ck] = (_Float16)vr[g][j];
    }
    __syncthreads();   // single barrier/iter (double buffer covers WAR hazard)

    // ---- issue next chunk's global loads; consumed only next iteration ----
    if (kc + 1 < nch) {
      const float* kp = Kg + kvbase + (size_t)(kb + 64) * D_;
      const float* vp = Vg + kvbase + (size_t)(kb + 64) * D_;
#pragma unroll
      for (int j = 0; j < 4; ++j) {
        kr[j] = *(const v4f*)(kp + 4 * j);
        vr[j] = *(const v4f*)(vp + 4 * j);
      }
    }

    // ---- S = (Q*log2e/sqrt(D)) * K^T : four 16x16 tiles over 64 keys ----
    v8f S[4];
#pragma unroll
    for (int g = 0; g < 4; ++g) S[g] = {};
#pragma unroll
    for (int c = 0; c < 2; ++c) {
#pragma unroll
      for (int g = 0; g < 4; ++g) {
        v16h Bk = lds_read16(bK + (g * 16 + lm) * KSTR + c * 32 + 16 * h);
        S[g] = wmma16x16x32(Qa[c], Bk, S[g]);
      }
    }

    // ---- length mask: only the (block-uniform) partial chunk needs it ----
    if (kb + 64 > vlen) {
      const float NEG = -1e30f;
#pragma unroll
      for (int g = 0; g < 4; ++g) {
        const bool ok = (kb + g * 16 + lm) < vlen;
#pragma unroll
        for (int i = 0; i < 8; ++i) S[g][i] = ok ? S[g][i] : NEG;
      }
    }

    // ---- online softmax (base-2 domain); row r = i + 8h ----
#pragma unroll
    for (int i = 0; i < 8; ++i) {
      float t = fmaxf(fmaxf(S[0][i], S[1][i]), fmaxf(S[2][i], S[3][i]));
      t = maxbcast16(t);
      float mn = fmaxf(mrow[i], t);
      corr8[i] = EXP2(mrow[i] - mn);
      mrow[i]  = mn;
#pragma unroll
      for (int g = 0; g < 4; ++g) S[g][i] = EXP2(S[g][i] - mn);
#pragma unroll
      for (int t4 = 0; t4 < 4; ++t4) Oacc[t4][i] *= corr8[i];
    }

    // ---- P (C-layout) -> A-layout via wave-private LDS bounce ----
    _Float16* pw = sP + wave * (16 * PSTR);
#pragma unroll
    for (int i = 0; i < 8; ++i) {
      _Float16* base = pw + (i + 8 * h) * PSTR;
#pragma unroll
      for (int g = 0; g < 4; ++g) base[g * 16 + lm] = (_Float16)S[g][i];
    }
    __builtin_amdgcn_wave_barrier();   // LDS in-order per wave; stop reordering
    asm volatile("" ::: "memory");
    v16h Pa[2];
#pragma unroll
    for (int c = 0; c < 2; ++c) {
      const _Float16* pr = pw + lm * PSTR + c * 32;
      v8h r0 = *(const v8h*)(pr + 16 * h);
      v8h r1 = *(const v8h*)(pr + 16 * h + 8);
#pragma unroll
      for (int j = 0; j < 8; ++j) { Pa[c][j] = r0[j]; Pa[c][8 + j] = r1[j]; }
    }

    // ---- row sums via WMMA: P * ones -> broadcast sums in C-layout ----
    v8f sums = {};
    sums = wmma16x16x32(Pa[0], onesB, sums);
    sums = wmma16x16x32(Pa[1], onesB, sums);
#pragma unroll
    for (int i = 0; i < 8; ++i) lrow[i] = lrow[i] * corr8[i] + sums[i];

    // ---- O += P * V ----
#pragma unroll
    for (int t4 = 0; t4 < 4; ++t4) {
#pragma unroll
      for (int c = 0; c < 2; ++c) {
        v16h Vb = lds_read16(bVt + (t4 * 16 + lm) * VSTR + c * 32 + 16 * h);
        Oacc[t4] = wmma16x16x32(Pa[c], Vb, Oacc[t4]);
      }
    }
  }

  // ---- epilogue: divide by l, store fp32 coalesced ----
#pragma unroll
  for (int i = 0; i < 8; ++i) {
    const int r = i + 8 * h;
    const float inv = 1.0f / lrow[i];
    float* orow = Og + ((size_t)b * SQ_ + q0 + r) * D_;
#pragma unroll
    for (int t4 = 0; t4 < 4; ++t4) orow[t4 * 16 + lm] = Oacc[t4][i] * inv;
  }
}

extern "C" void kernel_launch(void* const* d_in, const int* in_sizes, int n_in,
                              void* d_out, int out_size, void* d_ws, size_t ws_size,
                              hipStream_t stream) {
  (void)in_sizes; (void)n_in; (void)out_size; (void)d_ws; (void)ws_size;
  const float* Q  = (const float*)d_in[0];
  const float* K  = (const float*)d_in[1];
  const float* V  = (const float*)d_in[2];
  const int*   VL = (const int*)d_in[3];
  float* O = (float*)d_out;
  dim3 grid(B_ * (SQ_ / 128));   // 256 workgroups, one batch slice each
  dim3 block(256);               // 8 waves of 32
  fa_wmma_kernel<<<grid, block, 0, stream>>>(Q, K, V, VL, O);
}
